// HeteroHAN_11751030522362
// MI455X (gfx1250) — matrix operand order repaired and checked by reading, not compiled, not run on hardware
//
#include <hip/hip_runtime.h>
#include <hip/hip_bf16.h>
#include <math.h>

#define NN 100000
#define CC 256
#define HH 8
#define DD 32
#define SEMD 128
#define NEG_SLOPE 0.2f
#define TPB 256

typedef __attribute__((ext_vector_type(16))) __bf16 v16bf;
typedef __attribute__((ext_vector_type(8)))  float  v8f;

union BF16x16 { unsigned short u[16]; v16bf v; };

__device__ inline unsigned short f2bf(float f) {
  unsigned u = __float_as_uint(f);
  u += 0x7FFFu + ((u >> 16) & 1u);          // round-to-nearest-even bf16
  return (unsigned short)(u >> 16);
}
// monotone float -> uint map for atomicMax-based segment max
__device__ inline unsigned f2ord(float f) {
  unsigned u = __float_as_uint(f);
  return (u & 0x80000000u) ? ~u : (u | 0x80000000u);
}
__device__ inline float ord2f(unsigned m) {
  unsigned u = (m & 0x80000000u) ? (m & 0x7FFFFFFFu) : ~m;
  return __uint_as_float(u);
}

__global__ void fill_f32(float* p, float v, size_t n) {
  size_t i = (size_t)blockIdx.x * blockDim.x + threadIdx.x;
  if (i < n) p[i] = v;
}
__global__ void fill_u32(unsigned* p, unsigned v, size_t n) {
  size_t i = (size_t)blockIdx.x * blockDim.x + threadIdx.x;
  if (i < n) p[i] = v;
}

// ---------------- h = x @ W^T  (N x 256) * (256 x 32) via WMMA bf16 --------
__global__ void __launch_bounds__(TPB) gemm_h_kernel(const float* __restrict__ x,
                                                     const float* __restrict__ W,
                                                     float* __restrict__ h) {
  __shared__ unsigned short Wt[CC * DD];            // [k][n] bf16, 16KB
  int t = threadIdx.x;
  for (int idx = t; idx < CC * DD; idx += TPB) {
    int k = idx >> 5, n = idx & 31;
    Wt[idx] = f2bf(W[n * CC + k]);
  }
  __syncthreads();
  int wave = t >> 5, lane = t & 31;
  int tile = blockIdx.x * (TPB / 32) + wave;
  if (tile >= NN / 16) return;
  int row0 = tile * 16;
  int lm = lane & 15, lh = lane >> 4;
  v8f zero = {};
  v8f acc0 = zero, acc1 = zero;
  const float* xrow = x + (size_t)(row0 + lm) * CC;
  for (int kt = 0; kt < CC / 32; ++kt) {
    const float* p0 = xrow + kt * 32 + 8 * lh;      // A frag: K = kt*32 + 16*(j>>3) + 8*lh + (j&7)
    __builtin_prefetch(p0 + 32, 0, 1);
    BF16x16 a;
#pragma unroll
    for (int j = 0; j < 8; ++j) { a.u[j] = f2bf(p0[j]); a.u[j + 8] = f2bf(p0[16 + j]); }
    int K = kt * 32 + 8 * lh;                       // B frag: N = lm, K = K + 16*(j>>3) + (j&7)
    const unsigned short* wp = &Wt[K * DD + lm];
    BF16x16 b0, b1;
#pragma unroll
    for (int j = 0; j < 16; ++j) { b0.u[j] = wp[(j + (j & 8)) * DD]; b1.u[j] = wp[(j + (j & 8)) * DD + 16]; }
    acc0 = __builtin_amdgcn_wmma_f32_16x16x32_bf16(false, a.v, false, b0.v, (short)0, acc0, false, false);
    acc1 = __builtin_amdgcn_wmma_f32_16x16x32_bf16(false, a.v, false, b1.v, (short)0, acc1, false, false);
  }
#pragma unroll
  for (int r = 0; r < 8; ++r) {                     // C/D: M = r + 8*lh, N = lm
    int row = row0 + r + 8 * lh;
    h[(size_t)row * DD + lm]      = acc0[r];
    h[(size_t)row * DD + 16 + lm] = acc1[r];
  }
}

// -------- per-node attention dots: d[i,h]=h_i·a_dst[h], s[i,h]=h_i·a_src[h] ----
__global__ void node_dots_kernel(const float* __restrict__ h, const float* __restrict__ att,
                                 float* __restrict__ ddot, float* __restrict__ sdot) {
  int i = blockIdx.x * TPB + threadIdx.x;
  if (i >= NN * HH) return;
  int node = i >> 3, hh = i & 7;
  const float* hr = h + (size_t)node * DD;
  const float* ad = att + hh * 2 * DD;
  const float* as = ad + DD;
  float d = 0.f, s = 0.f;
#pragma unroll
  for (int k = 0; k < DD; ++k) { float hv = hr[k]; d += hv * ad[k]; s += hv * as[k]; }
  ddot[i] = d; sdot[i] = s;
}

// ---------------- segment softmax: pass 1 (max) ----------------------------
__global__ void edge_max_kernel(const int* __restrict__ src, const int* __restrict__ dst, int E,
                                const float* __restrict__ ddot, const float* __restrict__ sdot,
                                unsigned* __restrict__ maxb) {
  int i = blockIdx.x * TPB + threadIdx.x;
  if (i >= E * HH) return;
  int e = i >> 3, hh = i & 7;
  int s = src[e], d = dst[e];
  float v = ddot[d * HH + hh] + sdot[s * HH + hh];
  v = v > 0.f ? v : NEG_SLOPE * v;
  atomicMax(&maxb[d * HH + hh], f2ord(v));
}

// ---------------- segment softmax: pass 2 (sum of exp) ---------------------
__global__ void edge_sum_kernel(const int* __restrict__ src, const int* __restrict__ dst, int E,
                                const float* __restrict__ ddot, const float* __restrict__ sdot,
                                const unsigned* __restrict__ maxb, float* __restrict__ sumb) {
  int i = blockIdx.x * TPB + threadIdx.x;
  if (i >= E * HH) return;
  int e = i >> 3, hh = i & 7;
  int s = src[e], d = dst[e];
  int idx = d * HH + hh;
  float v = ddot[idx] + sdot[s * HH + hh];
  v = v > 0.f ? v : NEG_SLOPE * v;
  atomicAdd(&sumb[idx], __expf(v - ord2f(maxb[idx])));
}

// -------- pass 3: Z[dst, h, :] += alpha * h[src]  (one wave32 per edge) ----
__global__ void __launch_bounds__(TPB) edge_agg_kernel(const int* __restrict__ src,
    const int* __restrict__ dst, int E,
    const float* __restrict__ ddot, const float* __restrict__ sdot,
    const unsigned* __restrict__ maxb, const float* __restrict__ sumb,
    const float* __restrict__ h, float* __restrict__ Z) {
  int gid = blockIdx.x * TPB + threadIdx.x;
  int e = gid >> 5, lane = gid & 31;
  if (e >= E) return;
  int s = src[e], d = dst[e];
  float a = 0.f;
  if (lane < HH) {
    int idx = d * HH + lane;
    float v = ddot[idx] + sdot[s * HH + lane];
    v = v > 0.f ? v : NEG_SLOPE * v;
    a = __expf(v - ord2f(maxb[idx])) / sumb[idx];
  }
  float hk = h[(size_t)s * DD + lane];
  float* zr = Z + (size_t)d * CC;
#pragma unroll
  for (int hh = 0; hh < HH; ++hh) {
    float al = __shfl(a, hh, 32);
    atomicAdd(&zr[hh * DD + lane], al * hk);      // 128B coalesced atomic burst, hits L2
  }
}

// ---------------- elementwise ELU in place ---------------------------------
__global__ void elu_kernel(float* __restrict__ Z, size_t n) {
  size_t i = (size_t)blockIdx.x * TPB + threadIdx.x;
  if (i < n) { float z = Z[i]; Z[i] = z > 0.f ? z : (__expf(z) - 1.f); }
}

// ------- score = sum_i tanh(Z_i @ fc_w^T + b) · q   via WMMA bf16 ----------
__global__ void __launch_bounds__(TPB) score_kernel(const float* __restrict__ Z,
    const float* __restrict__ fc_w, const float* __restrict__ fc_b,
    const float* __restrict__ q, float* __restrict__ score) {
  __shared__ unsigned short fcT[CC * SEMD];         // [k][j] bf16, 64KB
  int t = threadIdx.x;
  for (int idx = t; idx < CC * SEMD; idx += TPB) {
    int k = idx >> 7, j = idx & 127;
    fcT[idx] = f2bf(fc_w[j * CC + k]);
  }
  __syncthreads();
  int wave = t >> 5, lane = t & 31;
  int tile = blockIdx.x * (TPB / 32) + wave;
  int lm = lane & 15, lh = lane >> 4;
  float partial = 0.f;
  if (tile < NN / 16) {
    int row0 = tile * 16;
    v8f zero = {};
    v8f acc[SEMD / 16];
#pragma unroll
    for (int nt = 0; nt < SEMD / 16; ++nt) acc[nt] = zero;
    const float* zrow = Z + (size_t)(row0 + lm) * CC;
    for (int kt = 0; kt < CC / 32; ++kt) {
      const float* p0 = zrow + kt * 32 + 8 * lh;
      __builtin_prefetch(p0 + 32, 0, 1);
      BF16x16 a;
#pragma unroll
      for (int j = 0; j < 8; ++j) { a.u[j] = f2bf(p0[j]); a.u[j + 8] = f2bf(p0[16 + j]); }
      int K = kt * 32 + 8 * lh;
      const unsigned short* bp = &fcT[K * SEMD + lm];
#pragma unroll
      for (int nt = 0; nt < SEMD / 16; ++nt) {
        BF16x16 b;
#pragma unroll
        for (int j = 0; j < 16; ++j) b.u[j] = bp[(j + (j & 8)) * SEMD + nt * 16];
        acc[nt] = __builtin_amdgcn_wmma_f32_16x16x32_bf16(false, a.v, false, b.v, (short)0, acc[nt], false, false);
      }
    }
#pragma unroll
    for (int nt = 0; nt < SEMD / 16; ++nt) {
      int j = nt * 16 + lm;
      float bj = fc_b[j], qj = q[j];
#pragma unroll
      for (int r = 0; r < 8; ++r) partial += tanhf(acc[nt][r] + bj) * qj;
    }
  }
#pragma unroll
  for (int m = 16; m > 0; m >>= 1) partial += __shfl_xor(partial, m, 32);
  if (lane == 0 && tile < NN / 16) atomicAdd(score, partial);
}

// ---------------- final 2-way semantic softmax blend -----------------------
__global__ void combine_kernel(float* __restrict__ out, const float* __restrict__ Z2,
                               const float* __restrict__ scores, size_t n) {
  size_t i = (size_t)blockIdx.x * TPB + threadIdx.x;
  if (i >= n) return;
  float a = scores[0] * (1.0f / NN);
  float b = scores[1] * (1.0f / NN);
  float m = fmaxf(a, b);
  float e0 = __expf(a - m), e1 = __expf(b - m);
  float inv = 1.f / (e0 + e1);
  out[i] = (e0 * inv) * out[i] + (e1 * inv) * Z2[i];
}

extern "C" void kernel_launch(void* const* d_in, const int* in_sizes, int n_in,
                              void* d_out, int out_size, void* d_ws, size_t ws_size,
                              hipStream_t stream) {
  const float* x    = (const float*)d_in[0];
  const float* W    = (const float*)d_in[1];
  const float* att1 = (const float*)d_in[2];
  const float* att2 = (const float*)d_in[3];
  const float* fc_w = (const float*)d_in[4];
  const float* fc_b = (const float*)d_in[5];
  const float* q    = (const float*)d_in[6];
  const int*   ei1  = (const int*)d_in[7];
  const int*   ei2  = (const int*)d_in[8];
  const int E1 = in_sizes[7] / 2;
  const int E2 = in_sizes[8] / 2;

  // workspace layout (floats): h | ddot | sdot | maxb | sumb | Z2 | scores[2]
  float*    h      = (float*)d_ws;
  float*    ddot   = h + (size_t)NN * DD;
  float*    sdot   = ddot + (size_t)NN * HH;
  unsigned* maxb   = (unsigned*)(sdot + (size_t)NN * HH);
  float*    sumb   = (float*)(maxb + (size_t)NN * HH);
  float*    Z2     = sumb + (size_t)NN * HH;
  float*    scores = Z2 + (size_t)NN * CC;
  float*    Z1     = (float*)d_out;                 // Z1 accumulates directly in d_out

  const int tiles = NN / 16;                        // 6250, exact
  const int gemmBlocks = (tiles + (TPB / 32) - 1) / (TPB / 32);
  const size_t nNC = (size_t)NN * CC;
  const size_t nNH = (size_t)NN * HH;

  fill_f32<<<1, 32, 0, stream>>>(scores, 0.f, 2);
  gemm_h_kernel<<<gemmBlocks, TPB, 0, stream>>>(x, W, h);

  const float* atts[2]  = {att1, att2};
  const int*   eis[2]   = {ei1, ei2};
  const int    Es[2]    = {E1, E2};
  float*       Zs[2]    = {Z1, Z2};

  for (int L = 0; L < 2; ++L) {
    const int* src = eis[L];
    const int* dst = eis[L] + Es[L];
    const int  E   = Es[L];
    float*     Z   = Zs[L];
    node_dots_kernel<<<(int)((nNH + TPB - 1) / TPB), TPB, 0, stream>>>(h, atts[L], ddot, sdot);
    fill_u32<<<(int)((nNH + TPB - 1) / TPB), TPB, 0, stream>>>(maxb, 0x007FFFFFu /* ord(-inf) */, nNH);
    fill_f32<<<(int)((nNH + TPB - 1) / TPB), TPB, 0, stream>>>(sumb, 0.f, nNH);
    fill_f32<<<(int)((nNC + TPB - 1) / TPB), TPB, 0, stream>>>(Z, 0.f, nNC);
    edge_max_kernel<<<(E * HH + TPB - 1) / TPB, TPB, 0, stream>>>(src, dst, E, ddot, sdot, maxb);
    edge_sum_kernel<<<(E * HH + TPB - 1) / TPB, TPB, 0, stream>>>(src, dst, E, ddot, sdot, maxb, sumb);
    edge_agg_kernel<<<(int)(((size_t)E * 32 + TPB - 1) / TPB), TPB, 0, stream>>>(
        src, dst, E, ddot, sdot, maxb, sumb, h, Z);
    elu_kernel<<<(int)((nNC + TPB - 1) / TPB), TPB, 0, stream>>>(Z, nNC);
    score_kernel<<<gemmBlocks, TPB, 0, stream>>>(Z, fc_w, fc_b, q, scores + L);
  }
  combine_kernel<<<(int)((nNC + TPB - 1) / TPB), TPB, 0, stream>>>(Z1, Z2, scores, nNC);
}
